// GraphNetwork_57208964383480
// MI455X (gfx1250) — matrix operand order repaired and checked by reading, not compiled, not run on hardware
//
#include <hip/hip_runtime.h>
#include <hip/hip_fp16.h>
#include <math.h>

// ---------------------------------------------------------------------------
// CDNA5 / gfx1250 implementation of the heterogeneous graph attention net.
// GEMMs (per-type projection, z = lrelu(agg) @ W) run on v_wmma_f32_16x16x32_f16.
// Edge gather / segment-softmax / scatter stages are f32 scalar (L2-resident).
// ---------------------------------------------------------------------------

typedef __attribute__((ext_vector_type(16))) _Float16 v16h;
typedef __attribute__((ext_vector_type(8)))  float    v8f;

#define FK   2
#define DK   32
#define DIN  64

// ---- WMMA fragment loaders (layouts per CDNA5 ISA 7.12.2, wave32) ----------

// A-matrix 16x32 f16: lane L holds row M = L&15.  Per VGPR p (halves 2p,2p+1):
// lanes 0-15:  K = {0,1},{2,3},{4,5},{6,7},{16,17},{18,19},{20,21},{22,23}
// lanes 16-31: K = {8,9},...,{14,15},{24,25},...,{30,31}
__device__ __forceinline__ v16h load_a_frag(const _Float16* A, int lda) {
  const int lane = threadIdx.x & 31;
  const int m    = lane & 15;
  const int hi   = lane >> 4;
  v16h a;
#pragma unroll
  for (int p = 0; p < 8; ++p) {
    const int k = ((p < 4) ? (2 * p) : (8 + 2 * p)) + hi * 8;
    a[2 * p]     = A[m * lda + k];
    a[2 * p + 1] = A[m * lda + k + 1];
  }
  return a;
}

// B-matrix 32x16 f16, staged column-major in LDS (Bt[n][k]): lane L holds
// column N = L&15; halves v hold K = hi*8 + v (v<8), 16 + hi*8 + (v-8) (v>=8).
__device__ __forceinline__ v16h load_b_frag(const _Float16* Bt, int ldb) {
  const int lane = threadIdx.x & 31;
  const int n    = lane & 15;
  const int hi   = lane >> 4;
  v16h b;
#pragma unroll
  for (int v = 0; v < 16; ++v) b[v] = Bt[n * ldb + hi * 8 + ((v < 8) ? v : (v + 8))];
  return b;
}

// ---------------------------------------------------------------------------
// Projection: all_emb[t][k] = l2norm(lrelu(emb[t] @ Wtk[t][k]))  (64 -> 32)
// Block = 64 threads (2 waves). Wave w computes output columns [16w,16w+16).
// K = 64 -> two v_wmma_f32_16x16x32_f16 per wave.
// ---------------------------------------------------------------------------
__global__ void proj_kernel(const float* __restrict__ emb,
                            const float* __restrict__ Wtk2,  // [FK][64][32]
                            float* __restrict__ outEmb,      // [FK][n][32]
                            int n) {
  __shared__ _Float16 sA[16 * DIN];    // [16][64] row-major
  __shared__ _Float16 sBt[DK * DIN];   // [32][64] col-major (Bt[n][k]=W[k][n])
  __shared__ float    sC[16 * DK];
  __shared__ float    sNinv[16];
  const int tid  = threadIdx.x;
  const int kf   = blockIdx.y;
  const int row0 = blockIdx.x * 16;
  const float* Wk = Wtk2 + (size_t)kf * DIN * DK;

  for (int idx = tid; idx < 16 * DIN; idx += 64) {
    const int r = idx >> 6, c = idx & 63;
    sA[idx] = (_Float16)emb[(size_t)(row0 + r) * DIN + c];
  }
  for (int idx = tid; idx < DK * DIN; idx += 64) {
    const int nn = idx >> 6, kk = idx & 63;
    sBt[idx] = (_Float16)Wk[kk * DK + nn];
  }
  __syncthreads();

  const int wv = tid >> 5;  // which 16-column tile this wave owns
  v8f c = {};
#pragma unroll
  for (int k0 = 0; k0 < DIN; k0 += 32) {
    v16h a = load_a_frag(sA + k0, DIN);
    v16h b = load_b_frag(sBt + wv * 16 * DIN + k0, DIN);
    c = __builtin_amdgcn_wmma_f32_16x16x32_f16(false, a, false, b,
                                               (short)0, c, false, false);
  }
  {  // leaky-relu, D layout: c[v] -> (M=v+8*hi, N=lane&15)
    const int lane = tid & 31;
    const int nn   = wv * 16 + (lane & 15);
    const int hi   = lane >> 4;
#pragma unroll
    for (int v = 0; v < 8; ++v) {
      const float x = c[v];
      sC[(v + hi * 8) * DK + nn] = (x > 0.f) ? x : 0.2f * x;
    }
  }
  __syncthreads();
  if (tid < 16) {
    float ss = 0.f;
#pragma unroll
    for (int d = 0; d < DK; ++d) { const float x = sC[tid * DK + d]; ss += x * x; }
    sNinv[tid] = 1.f / fmaxf(sqrtf(ss), 1e-12f);
  }
  __syncthreads();
  for (int idx = tid; idx < 16 * DK; idx += 64) {
    const int r = idx >> 5, d = idx & 31;
    outEmb[((size_t)kf * n + row0 + r) * DK + d] = sC[idx] * sNinv[r];
  }
}

// ---------------------------------------------------------------------------
// Edge pass 1: scores + segment max.  e_sum = mean_k relu(au.a1 + ai.a2)
// mbuf init 0; scores are >= 0 so int atomicMax on raw bits is order-correct.
// ---------------------------------------------------------------------------
__global__ void edge_scores(const float* __restrict__ eS, const float* __restrict__ eD,
                            const int* __restrict__ u, const int* __restrict__ ii,
                            const float* __restrict__ atE,  // [FK][64]
                            float* __restrict__ esum, float* __restrict__ mbuf,
                            int E, int nS, int nD) {
  __shared__ float sAt[FK * 2 * DK];  // 128 floats
  const int tid = threadIdx.x;
  if (tid < FK * 2 * DK) sAt[tid] = atE[tid];
  __syncthreads();
  const int e = blockIdx.x * blockDim.x + tid;
  if (e >= E) return;
  const int us = u[e], is = ii[e];
  float acc = 0.f;
#pragma unroll
  for (int k = 0; k < FK; ++k) {
    const float* au = eS + ((size_t)k * nS + us) * DK;
    const float* ai = eD + ((size_t)k * nD + is) * DK;
    const float* a1 = sAt + k * 2 * DK;
    const float* a2 = a1 + DK;
    float d = 0.f;
#pragma unroll
    for (int t = 0; t < DK; ++t) d += au[t] * a1[t] + ai[t] * a2[t];
    acc += fmaxf(d, 0.f);
  }
  const float es = acc * (1.f / FK);
  esum[e] = es;
  atomicMax((int*)&mbuf[us], __float_as_int(es));
}

// Edge pass 2: w = exp(e - m[seg]); segment sum.
__global__ void edge_expsum(float* __restrict__ wbuf, const float* __restrict__ mbuf,
                            float* __restrict__ sbuf, const int* __restrict__ u, int E) {
  const int e = blockIdx.x * blockDim.x + threadIdx.x;
  if (e >= E) return;
  const int us = u[e];
  const float w = __expf(wbuf[e] - mbuf[us]);
  wbuf[e] = w;
  atomicAdd(&sbuf[us], w);
}

// Edge pass 3: agg[k][u] += (w/s[u]) * ai[k]
__global__ void edge_scatter(const float* __restrict__ wbuf, const float* __restrict__ sbuf,
                             const int* __restrict__ u, const int* __restrict__ ii,
                             const float* __restrict__ eD, float* __restrict__ agg,
                             int E, int nS, int nD) {
  const int e = blockIdx.x * blockDim.x + threadIdx.x;
  if (e >= E) return;
  const int us = u[e], is = ii[e];
  const float wn = wbuf[e] / sbuf[us];
#pragma unroll
  for (int k = 0; k < FK; ++k) {
    const float* ai = eD + ((size_t)k * nD + is) * DK;
    float* ag = agg + ((size_t)k * nS + us) * DK;
#pragma unroll
    for (int d = 0; d < DK; ++d) atomicAdd(&ag[d], wn * ai[d]);
  }
}

// ---------------------------------------------------------------------------
// z = lrelu(agg) @ W  (32x32, one WMMA per wave) ; rlog = tanh(z) . q
// ---------------------------------------------------------------------------
__global__ void zw_kernel(const float* __restrict__ agg, const float* __restrict__ W,
                          const float* __restrict__ qv, float* __restrict__ z,
                          float* __restrict__ rlog, int n) {
  __shared__ _Float16 sA[16 * DK];
  __shared__ _Float16 sBt[DK * DK];
  __shared__ float    sZ[16 * DK];
  __shared__ float    sQ[DK];
  const int tid  = threadIdx.x;
  const int kf   = blockIdx.y;
  const int row0 = blockIdx.x * 16;

  for (int idx = tid; idx < 16 * DK; idx += 64) {
    const int r = idx >> 5, c = idx & 31;
    float x = agg[((size_t)kf * n + row0 + r) * DK + c];
    sA[idx] = (_Float16)((x > 0.f) ? x : 0.2f * x);  // lrelu before GEMM
  }
  for (int idx = tid; idx < DK * DK; idx += 64) {
    const int nn = idx >> 5, kk = idx & 31;
    sBt[idx] = (_Float16)W[kk * DK + nn];
  }
  if (tid < DK) sQ[tid] = qv[tid];
  __syncthreads();

  const int wv = tid >> 5;
  v16h a = load_a_frag(sA, DK);
  v16h b = load_b_frag(sBt + wv * 16 * DK, DK);
  v8f c = {};
  c = __builtin_amdgcn_wmma_f32_16x16x32_f16(false, a, false, b,
                                             (short)0, c, false, false);
  {
    const int lane = tid & 31;
    const int nn   = wv * 16 + (lane & 15);
    const int hi   = lane >> 4;
#pragma unroll
    for (int v = 0; v < 8; ++v) sZ[(v + hi * 8) * DK + nn] = c[v];
  }
  __syncthreads();
  for (int idx = tid; idx < 16 * DK; idx += 64) {
    const int r = idx >> 5, d = idx & 31;
    z[((size_t)kf * n + row0 + r) * DK + d] = sZ[idx];
  }
  if (tid < 16) {
    float acc = 0.f;
#pragma unroll
    for (int d = 0; d < DK; ++d) acc += tanhf(sZ[tid * DK + d]) * sQ[d];
    rlog[(size_t)kf * n + row0 + tid] = acc;
  }
}

// ---------------------------------------------------------------------------
// ego update: emb[t] = l2norm(emb[t] + sum_j z[j] * softmax_k(rlog[j]))
// ---------------------------------------------------------------------------
__global__ void update_kernel(float* __restrict__ embT, const float* __restrict__ zAll,
                              const float* __restrict__ rlogAll,
                              int jb, int je, int n) {
  const int idx = blockIdx.x * blockDim.x + threadIdx.x;
  if (idx >= 2 * n) return;
  const int kf = (idx >= n) ? 1 : 0;
  const int nn = idx - kf * n;
  float* row = embT + ((size_t)kf * n + nn) * DK;
  float ego[DK];
#pragma unroll
  for (int d = 0; d < DK; ++d) ego[d] = row[d];
  for (int j = jb; j < je; ++j) {
    const float* rl = rlogAll + (size_t)j * 2 * 50000;
    const float l0 = rl[nn], l1 = rl[50000 + nn];
    const float mx = fmaxf(l0, l1);
    const float e0 = __expf(l0 - mx), e1 = __expf(l1 - mx);
    const float r  = ((kf == 0) ? e0 : e1) / (e0 + e1);
    const float* zr = zAll + (size_t)j * 2 * 50000 * DK + ((size_t)kf * 50000 + nn) * DK;
#pragma unroll
    for (int d = 0; d < DK; ++d) ego[d] += zr[d] * r;
  }
  float ss = 0.f;
#pragma unroll
  for (int d = 0; d < DK; ++d) ss += ego[d] * ego[d];
  const float ninv = 1.f / fmaxf(sqrtf(ss), 1e-12f);
#pragma unroll
  for (int d = 0; d < DK; ++d) row[d] = ego[d] * ninv;
}

// out[n][k*32+d] = emb[k][n][d]   (concat factor slices)
__global__ void out_kernel(const float* __restrict__ embT, float* __restrict__ out, int n) {
  const int idx = blockIdx.x * blockDim.x + threadIdx.x;
  if (idx >= n * 64) return;
  const int nn = idx >> 6;
  const int c  = idx & 63;
  out[idx] = embT[(((size_t)(c >> 5)) * n + nn) * DK + (c & 31)];
}

__global__ void fill_kernel(float* __restrict__ p, float v, unsigned int cnt) {
  const unsigned int i = blockIdx.x * blockDim.x + threadIdx.x;
  if (i < cnt) p[i] = v;
}

// ---------------------------------------------------------------------------
extern "C" void kernel_launch(void* const* d_in, const int* in_sizes, int n_in,
                              void* d_out, int out_size, void* d_ws, size_t ws_size,
                              hipStream_t stream) {
  (void)in_sizes; (void)n_in; (void)out_size; (void)ws_size;
  static const int NTb[8]  = {50000, 50000, 10000, 10000, 10000, 10000, 10000, 10000};
  static const int ECb[8]  = {800000, 800000, 300000, 300000, 300000, 300000, 300000, 300000};
  static const int SRCt[8] = {0, 1, 1, 1, 1, 1, 1, 1};
  static const int DSTt[8] = {1, 0, 2, 3, 4, 5, 6, 7};

  const float* emb[8]; const int* uu[8]; const int* vv[8];
  for (int t = 0; t < 8; ++t) emb[t] = (const float*)d_in[t];
  for (int e = 0; e < 8; ++e) { uu[e] = (const int*)d_in[8 + 2 * e]; vv[e] = (const int*)d_in[9 + 2 * e]; }
  const float* Wtk = (const float*)d_in[24];  // [8][2][64][32]
  const float* at  = (const float*)d_in[25];  // [8][2][64]
  const float* W   = (const float*)d_in[26];  // [32][32]
  const float* q   = (const float*)d_in[27];  // [8][32]

  size_t nodePre[8]; { size_t a = 0; for (int t = 0; t < 8; ++t) { nodePre[t] = a; a += NTb[t]; } }

  // workspace carve (~163 MB total)
  float* ws = (float*)d_ws;
  size_t o = 0;
  float* allEmb = ws + o; o += (size_t)FK * DK * 160000;        // 10.24M f
  float* zAll   = ws + o; o += (size_t)8 * FK * 50000 * DK;     // 25.6M f
  float* rlogA  = ws + o; o += (size_t)8 * FK * 50000;          // 0.8M f
  float* agg    = ws + o; o += (size_t)FK * 50000 * DK;         // 3.2M f
  float* wbuf   = ws + o; o += 800000;
  float* mbuf   = ws + o; o += 50000;
  float* sbuf   = ws + o; o += 50000;

  auto embOff = [&](int t) { return (size_t)FK * DK * nodePre[t]; };

  // per-type factorized projection (WMMA)
  for (int t = 0; t < 8; ++t)
    proj_kernel<<<dim3(NTb[t] / 16, FK), 64, 0, stream>>>(
        emb[t], Wtk + (size_t)t * FK * DIN * DK, allEmb + embOff(t), NTb[t]);

  for (int it = 0; it < 4; ++it) {
    for (int e = 0; e < 8; ++e) {
      const int ns = NTb[SRCt[e]], nd = NTb[DSTt[e]], E = ECb[e];
      const unsigned int aggN = (unsigned int)(FK * ns * DK);
      fill_kernel<<<(ns + 255) / 256, 256, 0, stream>>>(mbuf, 0.f, (unsigned int)ns);
      fill_kernel<<<(ns + 255) / 256, 256, 0, stream>>>(sbuf, 0.f, (unsigned int)ns);
      fill_kernel<<<(aggN + 255) / 256, 256, 0, stream>>>(agg, 0.f, aggN);
      edge_scores<<<(E + 255) / 256, 256, 0, stream>>>(
          allEmb + embOff(SRCt[e]), allEmb + embOff(DSTt[e]), uu[e], vv[e],
          at + (size_t)e * FK * 2 * DK, wbuf, mbuf, E, ns, nd);
      edge_expsum<<<(E + 255) / 256, 256, 0, stream>>>(wbuf, mbuf, sbuf, uu[e], E);
      edge_scatter<<<(E + 255) / 256, 256, 0, stream>>>(
          wbuf, sbuf, uu[e], vv[e], allEmb + embOff(DSTt[e]), agg, E, ns, nd);
      zw_kernel<<<dim3(ns / 16, FK), 64, 0, stream>>>(
          agg, W, q + (size_t)e * DK,
          zAll + (size_t)e * FK * 50000 * DK, rlogA + (size_t)e * FK * 50000, ns);
    }
    update_kernel<<<(2 * NTb[0] + 255) / 256, 256, 0, stream>>>(
        allEmb + embOff(0), zAll, rlogA, 0, 1, NTb[0]);
    update_kernel<<<(2 * NTb[1] + 255) / 256, 256, 0, stream>>>(
        allEmb + embOff(1), zAll, rlogA, 1, 8, NTb[1]);
  }

  float* outp = (float*)d_out;
  size_t oo = 0;
  for (int t = 0; t < 8; ++t) {
    out_kernel<<<((unsigned int)(NTb[t] * 64) + 255) / 256, 256, 0, stream>>>(
        allEmb + embOff(t), outp + oo, NTb[t]);
    oo += (size_t)NTb[t] * 64;
  }
}
